// VGATEncoder_48198122996379
// MI455X (gfx1250) — matrix operand order repaired and checked by reading, not compiled, not run on hardware
//
#include <hip/hip_runtime.h>
#include <stdint.h>
#include <math.h>

#define N_NODES   50000
#define N_EDGES   800000
#define N_ETOT    850000   // edges + self loops
#define F_IN      256
#define HID       64
#define HEADS     4
#define F1        256      // HEADS*HID
#define LAT       32
#define NEG_SLOPE 0.2f

typedef __bf16 bf16_t;
typedef bf16_t v16bf __attribute__((ext_vector_type(16)));
typedef float  v8f   __attribute__((ext_vector_type(8)));

static __device__ __forceinline__ unsigned short f2bf(float f) {
  unsigned int u = __float_as_uint(f);
  u += 0x7fffu + ((u >> 16) & 1u);           // round-to-nearest-even
  return (unsigned short)(u >> 16);
}

// Native non-returning f32 atomic add (guarantees global_atomic_add_f32,
// no CAS-loop fallback). Kernel-end s_endpgm implies wait-idle.
static __device__ __forceinline__ void atomic_add_f32(float* p, float v) {
  asm volatile("global_atomic_add_f32 %0, %1, off scope:SCOPE_DEV"
               :
               : "v"((unsigned long long)(uintptr_t)p), "v"(v)
               : "memory");
}

// Float max via sign-aware integer atomics (native VMEM atomics, exact).
static __device__ __forceinline__ void atomic_max_f32(float* p, float v) {
  if (v >= 0.0f) atomicMax((int*)p, __float_as_int(v));
  else           atomicMin((unsigned int*)p, __float_as_uint(v));
}

// ---------------------------------------------------------------- utilities
__global__ void fill_f32(float* __restrict__ p, float v, int n) {
  int t = blockIdx.x * blockDim.x + threadIdx.x;
  if (t < n) p[t] = v;
}

__global__ void build_edges(const int* __restrict__ ei, int* __restrict__ src,
                            int* __restrict__ dst, int E, int N) {
  int t = blockIdx.x * blockDim.x + threadIdx.x;
  if (t >= E + N) return;
  if (t < E) { src[t] = ei[t]; dst[t] = ei[E + t]; }
  else       { src[t] = t - E; dst[t] = t - E; }
}

__global__ void convert_bf16(const float* __restrict__ in, unsigned short* __restrict__ out, int n) {
  int t = blockIdx.x * blockDim.x + threadIdx.x;
  if (t < n) out[t] = f2bf(in[t]);
}

// WT[n*K + k] = W[k*M + n]  (bf16), so B fragments read contiguous K-pairs.
__global__ void transpose_convert(const float* __restrict__ W, unsigned short* __restrict__ WT,
                                  int K, int M) {
  int t = blockIdx.x * blockDim.x + threadIdx.x;
  if (t >= K * M) return;
  int k = t / M, n = t % M;
  WT[(size_t)n * K + k] = f2bf(W[t]);
}

// ------------------------------------------------- WMMA GEMM: C = A * B
// A: [M,K] bf16 row-major, BT: [Nout,K] bf16 (B transposed), C: [M,Nout] f32.
// One wave per 16x64 strip: the A fragment is loaded once per 32-deep K step
// and reused by 4 v_wmma_f32_16x16x32_bf16 against 4 B fragments.
__global__ __launch_bounds__(256)
void gemm_bf16_wmma(const unsigned short* __restrict__ A,
                    const unsigned short* __restrict__ BT,
                    float* __restrict__ C, int M, int K, int Nout) {
  int gwave = (blockIdx.x * blockDim.x + threadIdx.x) >> 5;
  int lane  = threadIdx.x & 31;
  int groupsN = Nout >> 6;                  // 64 output columns per wave
  int tileM = gwave / groupsN;
  int ng    = gwave % groupsN;
  if (tileM * 16 >= M) return;

  int l16  = lane & 15;
  int half = lane >> 4;
  const unsigned int* Arow = (const unsigned int*)(A + (size_t)(tileM * 16 + l16) * K);
  const unsigned int* Brow = (const unsigned int*)(BT + (size_t)(ng * 64 + l16) * K);
  int Kd2 = K >> 1;                         // row stride in dwords

  v8f acc[4] = {};
  for (int k0 = 0; k0 < K; k0 += 32) {
    union { unsigned int u[8]; v16bf v; } a;
#pragma unroll
    for (int vi = 0; vi < 8; ++vi) {
      // A 16x32 bf16 fragment: VGPR vi<4 -> K = 2vi + 8*half; vi>=4 -> +16
      int ka = (vi < 4) ? (2 * vi + 8 * half) : (2 * (vi - 4) + 16 + 8 * half);
      a.u[vi] = Arow[(k0 + ka) >> 1];
    }
#pragma unroll
    for (int j = 0; j < 4; ++j) {
      union { unsigned int u[8]; v16bf v; } b;
      const unsigned int* Br = Brow + (size_t)j * 16 * Kd2;
#pragma unroll
      for (int vi = 0; vi < 8; ++vi) {
        // B 32x16 bf16 fragment: VGPR vi<4 -> K = 2vi + 8*half; vi>=4 -> +16, N = lane%16
        b.u[vi] = Br[(k0 + ((vi < 4) ? (2 * vi + 8 * half) : (2 * (vi - 4) + 16 + 8 * half))) >> 1];
      }
      acc[j] = __builtin_amdgcn_wmma_f32_16x16x32_bf16(false, a.v, false, b.v,
                                                       (short)0, acc[j], false, false);
    }
  }
  // C fragment: VGPR r holds rows r (lanes 0-15) / r+8 (lanes 16-31), col = lane%16
#pragma unroll
  for (int j = 0; j < 4; ++j) {
#pragma unroll
    for (int r = 0; r < 8; ++r) {
      int row = tileM * 16 + r + 8 * half;
      C[(size_t)row * Nout + ng * 64 + j * 16 + l16] = acc[j][r];
    }
  }
}

// ------------------------------------------------- attention pipeline
__global__ void node_scores(const float* __restrict__ H, const float* __restrict__ a_s,
                            const float* __restrict__ a_d, float* __restrict__ es,
                            float* __restrict__ ed, int n_nodes, int heads, int C) {
  int t = blockIdx.x * blockDim.x + threadIdx.x;
  if (t >= n_nodes * heads) return;
  int n = t / heads, h = t % heads;
  const float* hr = H + (size_t)n * heads * C + h * C;
  const float* av = a_s + h * C;
  const float* dv = a_d + h * C;
  float s = 0.f, d = 0.f;
  for (int c = 0; c < C; ++c) { float x = hr[c]; s = fmaf(x, av[c], s); d = fmaf(x, dv[c], d); }
  es[t] = s; ed[t] = d;
}

__global__ void edge_logits(const int* __restrict__ src, const int* __restrict__ dst,
                            const float* __restrict__ es, const float* __restrict__ ed,
                            float* __restrict__ el, float* __restrict__ emax,
                            int Etot, int heads) {
  int t = blockIdx.x * blockDim.x + threadIdx.x;
  if (t >= Etot * heads) return;
  int e = t / heads, h = t % heads;
  int s = src[e], d = dst[e];
  float x = es[s * heads + h] + ed[d * heads + h];
  x = (x > 0.f) ? x : x * NEG_SLOPE;
  el[t] = x;
  atomic_max_f32(&emax[d * heads + h], x);
}

__global__ void edge_exp(const int* __restrict__ dst, float* __restrict__ el,
                         const float* __restrict__ emax, float* __restrict__ den,
                         int Etot, int heads) {
  int t = blockIdx.x * blockDim.x + threadIdx.x;
  if (t >= Etot * heads) return;
  int e = t / heads, h = t % heads;
  int d = dst[e];
  float ee = __expf(el[t] - emax[d * heads + h]);
  el[t] = ee;
  atomic_add_f32(&den[d * heads + h], ee);
}

__global__ void edge_alpha(const int* __restrict__ dst, const float* __restrict__ ee,
                           const float* __restrict__ den, float* __restrict__ alpha,
                           int Etot, int heads) {
  int t = blockIdx.x * blockDim.x + threadIdx.x;
  if (t >= Etot * heads) return;
  int e = t / heads, h = t % heads;
  alpha[t] = ee[t] / (den[dst[e] * heads + h] + 1e-16f);
}

// One thread per (edge, 4 channels): float4 gather of h[src], 4 native atomic adds.
__global__ void edge_scatter(const int* __restrict__ src, const int* __restrict__ dst,
                             const float* __restrict__ H, const float* __restrict__ alpha,
                             float* __restrict__ agg, int Etot, int heads, int C) {
  int chunks = (heads * C) >> 2;
  int t = blockIdx.x * blockDim.x + threadIdx.x;
  if (t >= Etot * chunks) return;
  int e = t / chunks, q = t % chunks;
  int h = (q * 4) / C;
  int s = src[e], d = dst[e];
  float a = alpha[(size_t)e * heads + h];
  float4 hv = ((const float4*)(H + (size_t)s * heads * C))[q];
  float* out = agg + (size_t)d * heads * C + q * 4;
  atomic_add_f32(out + 0, hv.x * a);
  atomic_add_f32(out + 1, hv.y * a);
  atomic_add_f32(out + 2, hv.z * a);
  atomic_add_f32(out + 3, hv.w * a);
}

// layer1 finish: h1 = relu(agg + b1) stored as bf16 (next GEMM input)
__global__ void node_finish_concat(const float* __restrict__ agg, const float* __restrict__ b,
                                   unsigned short* __restrict__ outbf, int total, int F) {
  int t = blockIdx.x * blockDim.x + threadIdx.x;
  if (t >= total) return;
  float v = agg[t] + b[t % F];
  outbf[t] = f2bf(v > 0.f ? v : 0.f);
}

// layer2 finish: h2 = relu(mean_heads(agg) + b2)  (f32)
__global__ void node_finish_mean(const float* __restrict__ agg, const float* __restrict__ b,
                                 float* __restrict__ out, int n_nodes, int heads, int C) {
  int t = blockIdx.x * blockDim.x + threadIdx.x;
  if (t >= n_nodes * C) return;
  int n = t / C, c = t % C;
  float s = 0.f;
  for (int h = 0; h < heads; ++h) s += agg[(size_t)n * heads * C + h * C + c];
  float v = s * (1.0f / heads) + b[c];
  out[t] = v > 0.f ? v : 0.f;
}

// tiny f32 linear for the 64->32 heads: Y[n,m] = sum_k X[n,k]*W[k,m]
__global__ void small_linear(const float* __restrict__ X, const float* __restrict__ W,
                             float* __restrict__ Y, int n_nodes, int K, int M) {
  int t = blockIdx.x * blockDim.x + threadIdx.x;
  if (t >= n_nodes * M) return;
  int n = t / M, m = t % M;
  const float* xr = X + (size_t)n * K;
  float s = 0.f;
  for (int k = 0; k < K; ++k) s = fmaf(xr[k], W[k * M + m], s);
  Y[t] = s;
}

__global__ void head_finish(const float* __restrict__ agg, const float* __restrict__ b,
                            float* __restrict__ out, int total, int C) {
  int t = blockIdx.x * blockDim.x + threadIdx.x;
  if (t >= total) return;
  out[t] = agg[t] + b[t % C];
}

// ---------------------------------------------------------------- host side
static inline int cdiv(long long a, long long b) { return (int)((a + b - 1) / b); }

extern "C" void kernel_launch(void* const* d_in, const int* in_sizes, int n_in,
                              void* d_out, int out_size, void* d_ws, size_t ws_size,
                              hipStream_t stream) {
  const float* x    = (const float*)d_in[0];
  const int*   ei   = (const int*)d_in[1];
  const float* W1   = (const float*)d_in[2];
  const float* a1s  = (const float*)d_in[3];
  const float* a1d  = (const float*)d_in[4];
  const float* b1   = (const float*)d_in[5];
  const float* W2   = (const float*)d_in[6];
  const float* a2s  = (const float*)d_in[7];
  const float* a2d  = (const float*)d_in[8];
  const float* b2   = (const float*)d_in[9];
  const float* Wm   = (const float*)d_in[10];
  const float* ams  = (const float*)d_in[11];
  const float* amd_ = (const float*)d_in[12];
  const float* bm   = (const float*)d_in[13];
  const float* Ws   = (const float*)d_in[14];
  const float* assv = (const float*)d_in[15];
  const float* asd  = (const float*)d_in[16];
  const float* bs   = (const float*)d_in[17];

  float* zm  = (float*)d_out;                    // [N,32]
  float* zs  = zm + (size_t)N_NODES * LAT;       // [N,32]
  float* aw1 = zs + (size_t)N_NODES * LAT;       // [Etot,4]
  float* aw2 = aw1 + (size_t)N_ETOT * HEADS;     // [Etot,4]
  float* awm = aw2 + (size_t)N_ETOT * HEADS;     // [Etot,1]
  float* aws = awm + (size_t)N_ETOT;             // [Etot,1]

  char* w = (char*)d_ws;
  auto take = [&](size_t bytes) -> void* {
    void* p = (void*)w; w += (bytes + 255) & ~(size_t)255; return p;
  };
  int*            srci  = (int*)take((size_t)N_ETOT * 4);
  int*            dsti  = (int*)take((size_t)N_ETOT * 4);
  unsigned short* actbf = (unsigned short*)take((size_t)N_NODES * F1 * 2); // x / h1 bf16
  unsigned short* w1t   = (unsigned short*)take((size_t)F_IN * F1 * 2);
  unsigned short* w2t   = (unsigned short*)take((size_t)F1 * F1 * 2);
  float*          Hbuf  = (float*)take((size_t)N_NODES * F1 * 4);   // GEMM out / z_pre
  float*          AGG   = (float*)take((size_t)N_NODES * F1 * 4);
  float*          H2    = (float*)take((size_t)N_NODES * HID * 4);
  float*          ES    = (float*)take((size_t)N_NODES * HEADS * 4);
  float*          ED    = (float*)take((size_t)N_NODES * HEADS * 4);
  float*          EMAX  = (float*)take((size_t)N_NODES * HEADS * 4);
  float*          DEN   = (float*)take((size_t)N_NODES * HEADS * 4);
  float*          EEB   = (float*)take((size_t)N_ETOT * HEADS * 4);

  const int B = 256;

  auto run_att = [&](const float* Hf, const float* a_s, const float* a_d,
                     int heads, int C, float* alpha_out) {
    int nh = N_NODES * heads;
    fill_f32<<<cdiv(nh, B), B, 0, stream>>>(EMAX, -INFINITY, nh);
    fill_f32<<<cdiv(nh, B), B, 0, stream>>>(DEN, 0.f, nh);
    fill_f32<<<cdiv((long long)N_NODES * heads * C, B), B, 0, stream>>>(AGG, 0.f, N_NODES * heads * C);
    node_scores<<<cdiv(nh, B), B, 0, stream>>>(Hf, a_s, a_d, ES, ED, N_NODES, heads, C);
    int eh = N_ETOT * heads;
    edge_logits<<<cdiv(eh, B), B, 0, stream>>>(srci, dsti, ES, ED, EEB, EMAX, N_ETOT, heads);
    edge_exp<<<cdiv(eh, B), B, 0, stream>>>(dsti, EEB, EMAX, DEN, N_ETOT, heads);
    edge_alpha<<<cdiv(eh, B), B, 0, stream>>>(dsti, EEB, DEN, alpha_out, N_ETOT, heads);
    edge_scatter<<<cdiv((long long)N_ETOT * heads * C / 4, B), B, 0, stream>>>(
        srci, dsti, Hf, alpha_out, AGG, N_ETOT, heads, C);
  };

  // edges (with self loops) and bf16 operand prep
  build_edges<<<cdiv(N_ETOT, B), B, 0, stream>>>(ei, srci, dsti, N_EDGES, N_NODES);
  convert_bf16<<<cdiv((long long)N_NODES * F_IN, B), B, 0, stream>>>(x, actbf, N_NODES * F_IN);
  transpose_convert<<<cdiv(F_IN * F1, B), B, 0, stream>>>(W1, w1t, F_IN, F1);
  transpose_convert<<<cdiv(F1 * F1, B), B, 0, stream>>>(W2, w2t, F1, F1);

  // one wave per 16x64 strip
  int gemm_waves  = (N_NODES / 16) * (F1 / 64);                 // 12500 waves
  int gemm_blocks = cdiv((long long)gemm_waves * 32, B);        // 1563 blocks

  // ---- layer 1: h = x @ W1 (WMMA), attention, h1 = relu(agg + b1) -> bf16
  gemm_bf16_wmma<<<gemm_blocks, B, 0, stream>>>(actbf, w1t, Hbuf, N_NODES, F_IN, F1);
  run_att(Hbuf, a1s, a1d, HEADS, HID, aw1);
  node_finish_concat<<<cdiv((long long)N_NODES * F1, B), B, 0, stream>>>(AGG, b1, actbf,
                                                                         N_NODES * F1, F1);

  // ---- layer 2: h = h1 @ W2 (WMMA), attention, h2 = relu(mean + b2)
  gemm_bf16_wmma<<<gemm_blocks, B, 0, stream>>>(actbf, w2t, Hbuf, N_NODES, F1, F1);
  run_att(Hbuf, a2s, a2d, HEADS, HID, aw2);
  node_finish_mean<<<cdiv((long long)N_NODES * HID, B), B, 0, stream>>>(AGG, b2, H2,
                                                                        N_NODES, HEADS, HID);

  // ---- mean head: z_pre = h2 @ Wm, attention, z_mean = agg + bm
  small_linear<<<cdiv((long long)N_NODES * LAT, B), B, 0, stream>>>(H2, Wm, Hbuf, N_NODES, HID, LAT);
  run_att(Hbuf, ams, amd_, 1, LAT, awm);
  head_finish<<<cdiv((long long)N_NODES * LAT, B), B, 0, stream>>>(AGG, bm, zm, N_NODES * LAT, LAT);

  // ---- log_std head: z_pre = h2 @ Ws, attention, z_log_std = agg + bs
  small_linear<<<cdiv((long long)N_NODES * LAT, B), B, 0, stream>>>(H2, Ws, Hbuf, N_NODES, HID, LAT);
  run_att(Hbuf, assv, asd, 1, LAT, aws);
  head_finish<<<cdiv((long long)N_NODES * LAT, B), B, 0, stream>>>(AGG, bs, zs, N_NODES * LAT, LAT);
}
